// VE_ViT_Block_86552180949172
// MI455X (gfx1250) — matrix operand order repaired and checked by reading, not compiled, not run on hardware
//
#include <hip/hip_runtime.h>
#include <math.h>

typedef __attribute__((ext_vector_type(16))) __bf16 v16bf;
typedef __attribute__((ext_vector_type(8)))  float  v8f;

#define B_    64
#define N_    196
#define D_    768
#define H_    8
#define HD_   96
#define NP_   224
#define ROWS_ 12544
#define BH_   512
#define EPS_  1e-5f
#define SCALE_ 0.1020620726159658f   // 96^-0.5

__device__ __forceinline__ unsigned short f2bf(float f) {
  unsigned u = __builtin_bit_cast(unsigned, f);
  u += 0x7FFFu + ((u >> 16) & 1u);
  return (unsigned short)(u >> 16);
}

__device__ __forceinline__ float bredsum(float v, float* red) {
  int t = threadIdx.x;
  red[t] = v; __syncthreads();
  for (int s = 128; s > 0; s >>= 1) { if (t < s) red[t] += red[t + s]; __syncthreads(); }
  float r = red[0]; __syncthreads(); return r;
}
__device__ __forceinline__ float bredmax(float v, float* red) {
  int t = threadIdx.x;
  red[t] = v; __syncthreads();
  for (int s = 128; s > 0; s >>= 1) { if (t < s) red[t] = fmaxf(red[t], red[t + s]); __syncthreads(); }
  float r = red[0]; __syncthreads(); return r;
}

// ---------------- f32 -> bf16 weight conversion ----------------
__global__ __launch_bounds__(256) void cvt_kernel(const float* __restrict__ in,
                                                  unsigned short* __restrict__ out, int n) {
  int i = blockIdx.x * 256 + threadIdx.x;
  if (i < n) out[i] = f2bf(in[i]);
}

// ---------------- patch embed: dw conv 16x16 s16 p1 + pointwise 3->768 ----------------
__global__ __launch_bounds__(256) void patch_kernel(const float* __restrict__ img,
    const float* __restrict__ dww, const float* __restrict__ dwb,
    const float* __restrict__ pww, const float* __restrict__ pwb,
    float* __restrict__ xemb) {
  __shared__ float red[256];
  __shared__ float hc[3];
  int r = blockIdx.x; int b = r / N_, n = r % N_;
  int py = n / 14, px = n % 14;
  int t = threadIdx.x;
  int ky = t >> 4, kx = t & 15;
  int iy = py * 16 - 1 + ky, ix = px * 16 - 1 + kx;
  bool inb = (iy >= 0 && iy < 222 && ix >= 0 && ix < 222);
  for (int c = 0; c < 3; ++c) {
    float v = inb ? img[(((long long)b * 3 + c) * 222 + iy) * 222 + ix] : 0.0f;
    float s = bredsum(v * dww[c * 256 + t], red);
    if (t == 0) hc[c] = s + dwb[c];
    __syncthreads();
  }
  float h0 = hc[0], h1 = hc[1], h2 = hc[2];
  long long base = (long long)r * D_;
  for (int d = t; d < D_; d += 256)
    xemb[base + d] = pwb[d] + h0 * pww[d * 3] + h1 * pww[d * 3 + 1] + h2 * pww[d * 3 + 2];
}

// ---------------- LayerNorm (+optional pos add, +optional row stats) ----------------
__global__ __launch_bounds__(256) void ln_kernel(const float* __restrict__ x,
    const float* __restrict__ pos, const float* __restrict__ gamma, const float* __restrict__ beta,
    float* __restrict__ x1out, unsigned short* __restrict__ lnout,
    float* __restrict__ maxn, float* __restrict__ aven) {
  __shared__ float red[256];
  int row = blockIdx.x, t = threadIdx.x;
  long long base = (long long)row * D_;
  float v[3]; float s = 0.0f;
  #pragma unroll
  for (int i = 0; i < 3; ++i) {
    int idx = t + i * 256;
    float f = x[base + idx];
    if (pos) f += pos[(long long)(row % N_) * D_ + idx];
    v[i] = f; s += f;
  }
  float mean = bredsum(s, red) * (1.0f / D_);
  float sq = 0.0f;
  #pragma unroll
  for (int i = 0; i < 3; ++i) { float d = v[i] - mean; sq += d * d; }
  float var = bredsum(sq, red) * (1.0f / D_);
  float rstd = rsqrtf(var + EPS_);
  float lmax = -INFINITY, lsum = 0.0f;
  #pragma unroll
  for (int i = 0; i < 3; ++i) {
    int idx = t + i * 256;
    float y = (v[i] - mean) * rstd * gamma[idx] + beta[idx];
    lnout[base + idx] = f2bf(y);
    if (x1out) x1out[base + idx] = v[i];
    lmax = fmaxf(lmax, y); lsum += y;
  }
  if (maxn) {
    float rm = bredmax(lmax, red);
    float rs = bredsum(lsum, red);
    if (t == 0) { maxn[row] = rm; aven[row] = rs * (1.0f / D_); }
  }
}

// ---------------- gate stats ----------------
__global__ __launch_bounds__(256) void bstats_kernel(const float* __restrict__ maxn,
    const float* __restrict__ aven, float* __restrict__ maxg, float* __restrict__ aveg) {
  __shared__ float red[256];
  int b = blockIdx.x, t = threadIdx.x;
  float mv = (t < N_) ? maxn[b * N_ + t] : -INFINITY;
  float av = (t < N_) ? aven[b * N_ + t] : 0.0f;
  float rm = bredmax(mv, red);
  float rs = bredsum(av, red);
  if (t == 0) { maxg[b] = rm; aveg[b] = rs * (1.0f / N_); }
}

__global__ __launch_bounds__(256) void s1_kernel(const float* __restrict__ maxn,
    const float* __restrict__ maxg, float* __restrict__ s1) {
  __shared__ float red[256];
  int b = blockIdx.x, t = threadIdx.x;
  float tv = (t < N_) ? maxg[b] * maxn[b * N_ + t] : -INFINITY;
  float m = bredmax(tv, red);
  float e = (t < N_) ? expf(tv - m) : 0.0f;
  float s = bredsum(e, red);
  if (t < N_) s1[b * N_ + t] = e / s;
}

__global__ __launch_bounds__(256) void s2gate_kernel(const float* __restrict__ aven,
    const float* __restrict__ aveg, const float* __restrict__ s1, float* __restrict__ gate) {
  __shared__ float red[256];
  int n = blockIdx.x, t = threadIdx.x;
  float u = (t < B_) ? aveg[t] * aven[t * N_ + n] : -INFINITY;
  float m = bredmax(u, red);
  float e = (t < B_) ? expf(u - m) : 0.0f;
  float s = bredsum(e, red);
  if (t < B_) gate[t * N_ + n] = s1[t * N_ + n] * (e / s);
}

// ---------------- qkv repack: gate v, add x_emb, per-head layouts ----------------
__global__ __launch_bounds__(256) void pack_kernel(const float* __restrict__ qkv,
    const float* __restrict__ xemb, const float* __restrict__ gate,
    unsigned short* __restrict__ qpv, unsigned short* __restrict__ kh,
    unsigned short* __restrict__ vt) {
  int r = blockIdx.x; int b = r / N_, n = r % N_;
  float g = gate[r];
  long long qbase = (long long)r * (3 * D_);
  long long ebase = (long long)r * D_;
  for (int d = threadIdx.x; d < D_; d += 256) {
    float q = qkv[qbase + d];
    float k = qkv[qbase + D_ + d];
    float v = qkv[qbase + 2 * D_ + d] * g + xemb[ebase + d];
    int h = d / HD_, hd = d % HD_;
    long long bh = (long long)b * H_ + h;
    long long o = (bh * N_ + n) * HD_ + hd;
    qpv[o] = f2bf(q + v);
    kh[o]  = f2bf(k);
    vt[(bh * HD_ + hd) * NP_ + n] = f2bf(v);
  }
}

// ---------------- row softmax: scores(f32, pitch 224) -> P(bf16, zero-padded) ----------------
__global__ __launch_bounds__(256) void softmax_kernel(const float* __restrict__ sc,
    unsigned short* __restrict__ P) {
  __shared__ float red[256];
  int row = blockIdx.x, t = threadIdx.x;
  long long base = (long long)row * NP_;
  float v = (t < N_) ? sc[base + t] * SCALE_ : -INFINITY;
  float m = bredmax(v, red);
  float e = (t < N_) ? expf(v - m) : 0.0f;
  float s = bredsum(e, red);
  if (t < NP_) P[base + t] = (t < N_) ? f2bf(e / s) : (unsigned short)0;
}

// ---------------- bf16 WMMA GEMM: C[M,Nc] = A[M,K] * W[Nc,K]^T (+bias,+resid,gelu) ----------------
// 256 threads = 8 waves (4M x 2N); block tile 128x128; wave tile 32x64 = 2x4 WMMA.
// OOB rows/cols are address-clamped (not zeroed): garbage only reaches outputs the
// epilogue guard discards, so staging is fully branchless. Global loads for the
// iteration issue before the first barrier (overlap trailing WMMAs); no
// loop-carried registers, so nothing spills.
#define BM 128
#define BN 128
#define BK 32
#define LP 40   // LDS row pitch (ushort): 80B stride -> conflict-free, 16B aligned

union Frag { uint2 u[4]; v16bf v; };

__global__ __launch_bounds__(256) void gemm_bf16_kernel(
    const unsigned short* __restrict__ A, int lda, long long aStride,
    const unsigned short* __restrict__ W, int ldw, long long wStride,
    float* __restrict__ C, unsigned short* __restrict__ Cb, int ldc, int ldcb,
    int cDiv, long long cOuter, long long cInner,
    const float* __restrict__ bias, const float* __restrict__ resid,
    int M, int Nc, int K, int doGelu) {
  __shared__ __align__(16) unsigned short lA[BM][LP];
  __shared__ __align__(16) unsigned short lB[BN][LP];
  const int t = threadIdx.x;
  const int lane = t & 31, wv = t >> 5;
  const int l15 = lane & 15, hi = lane >> 4;
  const int waveM = (wv >> 1) * 32, waveN = (wv & 1) * 64;
  const int bz = blockIdx.z;
  const unsigned short* Ab = A + (long long)bz * aStride;
  const unsigned short* Wb = W + (long long)bz * wStride;
  const long long cOff = (long long)(bz / cDiv) * cOuter + (long long)(bz % cDiv) * cInner;
  const int m0 = blockIdx.y * BM, n0 = blockIdx.x * BN;

  // staging addresses (row/col clamped -> always in-bounds, branchless)
  // 512 (row,seg) slots over 2 rounds: idx = t + r*256; row = idx>>2; seg = idx&3
  const unsigned short* aSrc0; const unsigned short* aSrc1;
  const unsigned short* wSrc0; const unsigned short* wSrc1;
  unsigned short* aDst0; unsigned short* aDst1;
  unsigned short* wDst0; unsigned short* wDst1;
  {
    int row0 = t >> 2, seg0 = t & 3;
    int idx1 = t + 256;
    int row1 = idx1 >> 2, seg1 = idx1 & 3;
    int gm0 = m0 + row0; if (gm0 > M - 1) gm0 = M - 1;
    int gm1 = m0 + row1; if (gm1 > M - 1) gm1 = M - 1;
    int gn0 = n0 + row0; if (gn0 > Nc - 1) gn0 = Nc - 1;
    int gn1 = n0 + row1; if (gn1 > Nc - 1) gn1 = Nc - 1;
    aSrc0 = Ab + (long long)gm0 * lda + seg0 * 8;
    aSrc1 = Ab + (long long)gm1 * lda + seg1 * 8;
    wSrc0 = Wb + (long long)gn0 * ldw + seg0 * 8;
    wSrc1 = Wb + (long long)gn1 * ldw + seg1 * 8;
    aDst0 = &lA[row0][seg0 * 8];
    aDst1 = &lA[row1][seg1 * 8];
    wDst0 = &lB[row0][seg0 * 8];
    wDst1 = &lB[row1][seg1 * 8];
  }

  v8f acc[2][4];
  v8f z = {0.f,0.f,0.f,0.f,0.f,0.f,0.f,0.f};
  #pragma unroll
  for (int i = 0; i < 2; ++i)
    #pragma unroll
    for (int j = 0; j < 4; ++j) acc[i][j] = z;

  for (int kt = 0; kt < K; kt += BK) {
    // global loads first: overlap other waves' trailing WMMAs from previous iter
    uint4 a0 = *(const uint4*)(aSrc0 + kt);
    uint4 a1 = *(const uint4*)(aSrc1 + kt);
    uint4 b0 = *(const uint4*)(wSrc0 + kt);
    uint4 b1 = *(const uint4*)(wSrc1 + kt);
    __syncthreads();
    *(uint4*)aDst0 = a0;
    *(uint4*)aDst1 = a1;
    *(uint4*)wDst0 = b0;
    *(uint4*)wDst1 = b1;
    __syncthreads();
    Frag fa[2], fb[4];
    #pragma unroll
    for (int ti = 0; ti < 2; ++ti) {
      // A 16x32 layout: lanes 0-15 K{0..7,16..23}, lanes 16-31 K{8..15,24..31}
      const unsigned short* p = &lA[waveM + ti * 16 + l15][hi * 8];
      fa[ti].u[0] = *(const uint2*)(p);
      fa[ti].u[1] = *(const uint2*)(p + 4);
      fa[ti].u[2] = *(const uint2*)(p + 16);
      fa[ti].u[3] = *(const uint2*)(p + 20);
    }
    #pragma unroll
    for (int tj = 0; tj < 4; ++tj) {
      // B 32x16 layout: lanes 0-15 K{0..7,16..23}, lanes 16-31 K{8..15,24..31}, column = lane&15
      const unsigned short* p = &lB[waveN + tj * 16 + l15][hi * 8];
      fb[tj].u[0] = *(const uint2*)(p);
      fb[tj].u[1] = *(const uint2*)(p + 4);
      fb[tj].u[2] = *(const uint2*)(p + 16);
      fb[tj].u[3] = *(const uint2*)(p + 20);
    }
    #pragma unroll
    for (int ti = 0; ti < 2; ++ti)
      #pragma unroll
      for (int tj = 0; tj < 4; ++tj)
        acc[ti][tj] = __builtin_amdgcn_wmma_f32_16x16x32_bf16(
            false, fa[ti].v, false, fb[tj].v, (short)0, acc[ti][tj], false, false);
  }

  #pragma unroll
  for (int ti = 0; ti < 2; ++ti)
    #pragma unroll
    for (int tj = 0; tj < 4; ++tj)
      #pragma unroll
      for (int j = 0; j < 8; ++j) {
        int m = m0 + waveM + ti * 16 + hi * 8 + j;
        int n = n0 + waveN + tj * 16 + l15;
        if (m < M && n < Nc) {
          float v = acc[ti][tj][j];
          if (bias)  v += bias[n];
          if (resid) v += resid[cOff + (long long)m * ldc + n];
          if (doGelu) v = 0.5f * v * (1.0f + erff(v * 0.70710678f));
          if (C)  C[cOff + (long long)m * ldc + n] = v;
          if (Cb) Cb[cOff + (long long)m * ldcb + n] = f2bf(v);
        }
      }
}

// ---------------- host ----------------
extern "C" void kernel_launch(void* const* d_in, const int* in_sizes, int n_in,
                              void* d_out, int out_size, void* d_ws, size_t ws_size,
                              hipStream_t stream) {
  const float* x     = (const float*)d_in[0];
  const float* xEem  = (const float*)d_in[1];
  const float* pos   = (const float*)d_in[2];
  const float* dww   = (const float*)d_in[3];
  const float* dwb   = (const float*)d_in[4];
  const float* pww   = (const float*)d_in[5];
  const float* pwb   = (const float*)d_in[6];
  const float* qkvw  = (const float*)d_in[7];
  const float* projw = (const float*)d_in[8];
  const float* projb = (const float*)d_in[9];
  const float* ln1s  = (const float*)d_in[10];
  const float* ln1b  = (const float*)d_in[11];
  const float* ln2s  = (const float*)d_in[12];
  const float* ln2b  = (const float*)d_in[13];
  const float* fc1w  = (const float*)d_in[14];
  const float* fc1b  = (const float*)d_in[15];
  const float* fc2w  = (const float*)d_in[16];
  const float* fc2b  = (const float*)d_in[17];
  float* out = (float*)d_out;

  char* wsb = (char*)d_ws;
  size_t off = 0;
  auto take = [&](size_t bytes) -> char* {
    char* p = wsb + off; off += (bytes + 255) & ~(size_t)255; return p;
  };

  unsigned short* qkvw_bf  = (unsigned short*)take((size_t)2304 * 768 * 2);
  unsigned short* projw_bf = (unsigned short*)take((size_t)768 * 768 * 2);
  unsigned short* fc1w_bf  = (unsigned short*)take((size_t)3072 * 768 * 2);
  unsigned short* fc2w_bf  = (unsigned short*)take((size_t)768 * 3072 * 2);
  float* xemb  = (float*)take((size_t)ROWS_ * D_ * 4);
  float* x1    = (float*)take((size_t)ROWS_ * D_ * 4);
  unsigned short* ln1bf = (unsigned short*)take((size_t)ROWS_ * D_ * 2);
  float* big   = (float*)take((size_t)ROWS_ * 3 * D_ * 4);   // qkv f32 / scores f32 / gelu bf16
  unsigned short* qpv = (unsigned short*)take((size_t)BH_ * N_ * HD_ * 2);
  unsigned short* kh  = (unsigned short*)take((size_t)BH_ * N_ * HD_ * 2);
  unsigned short* vt  = (unsigned short*)take((size_t)BH_ * HD_ * NP_ * 2);
  unsigned short* pmat= (unsigned short*)take((size_t)BH_ * N_ * NP_ * 2);
  unsigned short* acat= (unsigned short*)take((size_t)ROWS_ * D_ * 2);
  float* x2    = (float*)take((size_t)ROWS_ * D_ * 4);
  unsigned short* ln2bf = (unsigned short*)take((size_t)ROWS_ * D_ * 2);
  float* maxn = (float*)take((size_t)ROWS_ * 4);
  float* aven = (float*)take((size_t)ROWS_ * 4);
  float* maxg = (float*)take(256);
  float* aveg = (float*)take(256);
  float* s1   = (float*)take((size_t)ROWS_ * 4);
  float* gate = (float*)take((size_t)ROWS_ * 4);
  float* scores = big;
  unsigned short* hgelu = (unsigned short*)big;

  // weight conversion f32 -> bf16
  auto cvt = [&](const float* in, unsigned short* o, int n) {
    cvt_kernel<<<dim3((n + 255) / 256), dim3(256), 0, stream>>>(in, o, n);
  };
  cvt(qkvw,  qkvw_bf,  2304 * 768);
  cvt(projw, projw_bf, 768 * 768);
  cvt(fc1w,  fc1w_bf,  3072 * 768);
  cvt(fc2w,  fc2w_bf,  768 * 3072);

  // patch embed
  patch_kernel<<<dim3(ROWS_), dim3(256), 0, stream>>>(xEem, dww, dwb, pww, pwb, xemb);

  // x1 = x + pos ; ln1 -> bf16 ; row stats for gate
  ln_kernel<<<dim3(ROWS_), dim3(256), 0, stream>>>(x, pos, ln1s, ln1b, x1, ln1bf, maxn, aven);

  bstats_kernel<<<dim3(B_), dim3(256), 0, stream>>>(maxn, aven, maxg, aveg);
  s1_kernel<<<dim3(B_), dim3(256), 0, stream>>>(maxn, maxg, s1);
  s2gate_kernel<<<dim3(N_), dim3(256), 0, stream>>>(aven, aveg, s1, gate);

  // qkv GEMM: (12544x768) x (2304x768)^T -> f32
  gemm_bf16_kernel<<<dim3(2304 / BN, ROWS_ / BM, 1), dim3(256), 0, stream>>>(
      ln1bf, 768, 0, qkvw_bf, 768, 0,
      big, nullptr, 2304, 0, 1, 0, 0,
      nullptr, nullptr, ROWS_, 2304, 768, 0);

  hipMemsetAsync(vt, 0, (size_t)BH_ * HD_ * NP_ * 2, stream);
  pack_kernel<<<dim3(ROWS_), dim3(256), 0, stream>>>(big, xemb, gate, qpv, kh, vt);

  // scores: batched (196x96) x (196x96)^T -> f32, pitch 224 (reuses `big`)
  gemm_bf16_kernel<<<dim3((N_ + BN - 1) / BN, (N_ + BM - 1) / BM, BH_), dim3(256), 0, stream>>>(
      qpv, HD_, (long long)N_ * HD_, kh, HD_, (long long)N_ * HD_,
      scores, nullptr, NP_, 0, 1, (long long)N_ * NP_, 0,
      nullptr, nullptr, N_, N_, HD_, 0);

  softmax_kernel<<<dim3(BH_ * N_), dim3(256), 0, stream>>>(scores, pmat);

  // attn @ V: batched (196x224) x (96x224)^T -> bf16 written into (B,N,D) concat layout
  gemm_bf16_kernel<<<dim3((HD_ + BN - 1) / BN, (N_ + BM - 1) / BM, BH_), dim3(256), 0, stream>>>(
      pmat, NP_, (long long)N_ * NP_, vt, NP_, (long long)HD_ * NP_,
      nullptr, acat, D_, D_, H_, (long long)N_ * D_, HD_,
      nullptr, nullptr, N_, HD_, NP_, 0);

  // proj + bias + residual(x1) -> x2 (f32)
  gemm_bf16_kernel<<<dim3(D_ / BN, ROWS_ / BM, 1), dim3(256), 0, stream>>>(
      acat, 768, 0, projw_bf, 768, 0,
      x2, nullptr, 768, 0, 1, 0, 0,
      projb, x1, ROWS_, 768, 768, 0);

  // ln2 -> bf16
  ln_kernel<<<dim3(ROWS_), dim3(256), 0, stream>>>(x2, nullptr, ln2s, ln2b,
                                                   nullptr, ln2bf, nullptr, nullptr);

  // fc1 + bias + GELU -> bf16 (reuses `big`)
  gemm_bf16_kernel<<<dim3(3072 / BN, ROWS_ / BM, 1), dim3(256), 0, stream>>>(
      ln2bf, 768, 0, fc1w_bf, 768, 0,
      nullptr, hgelu, 3072, 3072, 1, 0, 0,
      fc1b, nullptr, ROWS_, 3072, 768, 1);

  // fc2 + bias + residual(x2) -> out (f32)
  gemm_bf16_kernel<<<dim3(D_ / BN, ROWS_ / BM, 1), dim3(256), 0, stream>>>(
      hgelu, 3072, 0, fc2w_bf, 3072, 0,
      out, nullptr, 768, 0, 1, 0, 0,
      fc2b, x2, ROWS_, 768, 3072, 0);

  (void)in_sizes; (void)n_in; (void)out_size; (void)ws_size;
}
